// Decoder_48490180772318
// MI455X (gfx1250) — matrix operand order repaired and checked by reading, not compiled, not run on hardware
//
#include <hip/hip_runtime.h>

#define B_   8
#define S_   512
#define H_   768
#define V_   35004
#define J_   30
#define KK_  9
#define H3_  2304
#define VP_  35008
#define NT_V 2188      /* VP_/16 vocab N-tiles   */
#define NT_G 144       /* H3_/16 GRU   N-tiles   */
#define KSTEPS 24      /* H_/32 WMMA K steps     */

typedef __attribute__((ext_vector_type(16))) _Float16 v16h;
typedef __attribute__((ext_vector_type(8)))  _Float16 v8h;
typedef __attribute__((ext_vector_type(8)))  float    v8f;

// ---- monotone float <-> ordered-uint keys (for deterministic atomicMax) ----
__device__ __forceinline__ unsigned fkey(float f){
  unsigned u = __float_as_uint(f);
  return (u & 0x80000000u) ? ~u : (u | 0x80000000u);
}
__device__ __forceinline__ float funkey(unsigned k){
  return __uint_as_float((k & 0x80000000u) ? (k ^ 0x80000000u) : ~k);
}

// ---- A fragment: 16-bit A 16x32 layout (lane M=L&15, kq=8*(L>=16);
//      elems 0..7 -> k=kq+e, elems 8..15 -> k=16+kq+(e-8)) ----
__device__ __forceinline__ v16h load_a_frag(const _Float16* __restrict__ base, int k0){
  int lane = threadIdx.x & 31;
  int m  = lane & 15;
  int kq = (lane & 16) ? 8 : 0;
  const _Float16* p = base + m * H_ + k0 + kq;
  v8h lo = *(const v8h*)p;
  v8h hi = *(const v8h*)(p + 16);
  v16h a;
#pragma unroll
  for (int i = 0; i < 8; ++i){ a[i] = lo[i]; a[i + 8] = hi[i]; }
  return a;
}

// ---- B fragment from pre-packed layout: one aligned 32B load per lane ----
__device__ __forceinline__ v16h load_b_packed(const _Float16* __restrict__ packed, int tile, int q){
  int lane = threadIdx.x & 31;
  return *(const v16h*)(packed + ((((size_t)tile * KSTEPS + q) * 32) + lane) * 16);
}

// ---- pack f32 [nrows, H] weights into B-fragment-linear f16 layout ----
// chunk id = (tile*KSTEPS + q)*32 + lane ; lane covers N=tile*16+(lane&15),
// K = q*32 + ((lane&16)?8:0) + e for e<8, q*32 + 16 + ((lane&16)?8:0) + (e-8) for e>=8.
__global__ void pack_b_kernel(const float* __restrict__ src, _Float16* __restrict__ dst,
                              int nrows, int ntiles){
  size_t idx = (size_t)blockIdx.x * 256 + threadIdx.x;
  size_t total = (size_t)ntiles * KSTEPS * 32;
  if (idx >= total) return;
  int lane = (int)(idx & 31);
  size_t tq = idx >> 5;
  int q    = (int)(tq % KSTEPS);
  int tile = (int)(tq / KSTEPS);
  int row  = tile * 16 + (lane & 15);
  if (row >= nrows) row = nrows - 1;
  int kb = (lane & 16) ? 8 : 0;
  const float* s = src + (size_t)row * H_ + q * 32 + kb;
  _Float16* d = dst + idx * 16;
#pragma unroll
  for (int e = 0; e < 16; ++e) d[e] = (_Float16)s[e < 8 ? e : e + 8];
}

// ---- initial state: h from hidden[0], w(slot 0) from decoder_input; zero-pad rows 8..15 ----
__global__ void init_state_kernel(const float* __restrict__ hidden, const float* __restrict__ dec_in,
                                  float* __restrict__ h32, _Float16* __restrict__ h16,
                                  float* __restrict__ w32a, _Float16* __restrict__ w16a,
                                  _Float16* __restrict__ w16b){
  int idx = blockIdx.x * 256 + threadIdx.x;      // 0 .. 16*H_-1
  int m = idx / H_, hh = idx % H_;
  if (m < B_){
    float hv = hidden[m * H_ + hh];
    h32[idx] = hv;            h16[idx] = (_Float16)hv;
    float wv = dec_in[((size_t)m * J_ + 0) * H_ + hh];
    w32a[idx] = wv;           w16a[idx] = (_Float16)wv;
    w16b[idx] = (_Float16)0.f;
  } else {
    h16[idx]  = (_Float16)0.f;
    w16a[idx] = (_Float16)0.f;
    w16b[idx] = (_Float16)0.f;
  }
}

// ---- GRU matmuls: gi = w @ w_ih^T, gh = h @ w_hh^T (biases added later) ----
__global__ void gru_mm_kernel(const _Float16* __restrict__ w16, const _Float16* __restrict__ h16,
                              const _Float16* __restrict__ wihP, const _Float16* __restrict__ whhP,
                              float* __restrict__ gi, float* __restrict__ gh){
  const _Float16* A  = blockIdx.y ? h16  : w16;
  const _Float16* Bp = blockIdx.y ? whhP : wihP;
  float* out         = blockIdx.y ? gh   : gi;
  int wave = threadIdx.x >> 5;
  int lane = threadIdx.x & 31;
  int tile = blockIdx.x * 4 + wave;              // 0..143
  v8f c = {};
  for (int q = 0; q < KSTEPS; ++q){
    v16h a = load_a_frag(A, q * 32);
    v16h b = load_b_packed(Bp, tile, q);
    c = __builtin_amdgcn_wmma_f32_16x16x32_f16(false, a, false, b, (short)0, c, false, false);
  }
  if (lane < 16){
    int n = tile * 16 + lane;
#pragma unroll
    for (int r = 0; r < 8; ++r) out[(size_t)r * H3_ + n] = c[r];   // D: VGPR r -> M=r, N=lane
  }
}

// ---- gates + h update + prep next-step w (ping-pong buffer) ----
__global__ void gru_gates_kernel(const float* __restrict__ gi, const float* __restrict__ gh,
                                 const float* __restrict__ b_ih, const float* __restrict__ b_hh,
                                 float* __restrict__ h32, _Float16* __restrict__ h16,
                                 const float* __restrict__ dec_in, const float* __restrict__ embed,
                                 const int* __restrict__ teacher,
                                 float* __restrict__ w32n, _Float16* __restrict__ w16n, int t){
  int idx = blockIdx.x * 256 + threadIdx.x;      // 0 .. B_*H_-1
  int b = idx / H_, hh = idx % H_;
  float ir  = gi[b * H3_ + hh]           + b_ih[hh];
  float iz  = gi[b * H3_ + H_ + hh]      + b_ih[H_ + hh];
  float in_ = gi[b * H3_ + 2 * H_ + hh]  + b_ih[2 * H_ + hh];
  float hr  = gh[b * H3_ + hh]           + b_hh[hh];
  float hz  = gh[b * H3_ + H_ + hh]      + b_hh[H_ + hh];
  float hn  = gh[b * H3_ + 2 * H_ + hh]  + b_hh[2 * H_ + hh];
  float r = 1.f / (1.f + expf(-(ir + hr)));
  float z = 1.f / (1.f + expf(-(iz + hz)));
  float n = tanhf(in_ + r * hn);
  float hnew = (1.f - z) * n + z * h32[idx];
  h32[idx] = hnew;
  h16[idx] = (_Float16)hnew;
  int tn = t + 1;
  if (tn < J_ * KK_){
    int j = tn / KK_, k = tn % KK_;
    float wv;
    if (k == 0) wv = dec_in[((size_t)b * J_ + j) * H_ + hh];
    else {
      int tok = teacher[(b * J_ + j) * KK_ + (k - 1)];
      wv = embed[(size_t)tok * H_ + hh];
    }
    w32n[idx] = wv;
    w16n[idx] = (_Float16)wv;
  }
}

// ---- attention softmax + context + p_gen; also resets vocab-softmax state ----
__global__ void attn_kernel(const float* __restrict__ enc, const int* __restrict__ x,
                            const float* __restrict__ h32, const float* __restrict__ w32,
                            const float* __restrict__ wg_w, const float* __restrict__ wg_b,
                            float* __restrict__ ahist, float* __restrict__ pgen,
                            unsigned* __restrict__ rowmaxkey, float* __restrict__ partial){
  int b = blockIdx.x;
  int tid = threadIdx.x;                         // 0..511 == s
  __shared__ float sl[S_];
  __shared__ float sctx[H_];
  __shared__ float red[16];

  if (tid == 0) rowmaxkey[b] = 0u;               // key for most-negative
  if (tid < 32) partial[b * 32 + tid] = 0.f;

  const float* er = enc + ((size_t)b * S_ + tid) * H_;
  float acc = 0.f;
  for (int i = 0; i < H_; ++i) acc += er[i] * h32[b * H_ + i];
  if (x[b * S_ + tid] == 0) acc = -1e9f;         // PAD mask
  sl[tid] = acc;
  __syncthreads();

  float m = sl[tid];
  for (int o = 16; o > 0; o >>= 1) m = fmaxf(m, __shfl_xor(m, o, 32));
  if ((tid & 31) == 0) red[tid >> 5] = m;
  __syncthreads();
  if (tid == 0){ float mm = red[0]; for (int i = 1; i < 16; ++i) mm = fmaxf(mm, red[i]); red[0] = mm; }
  __syncthreads();
  float smax = red[0];
  __syncthreads();

  float e = expf(sl[tid] - smax);
  float s = e;
  for (int o = 16; o > 0; o >>= 1) s += __shfl_xor(s, o, 32);
  if ((tid & 31) == 0) red[tid >> 5] = s;
  __syncthreads();
  if (tid == 0){ float ss = 0.f; for (int i = 0; i < 16; ++i) ss += red[i]; red[0] = ss; }
  __syncthreads();
  float a = e / red[0];
  sl[tid] = a;
  ahist[b * S_ + tid] = a;
  __syncthreads();

  for (int hh = tid; hh < H_; hh += S_){
    float c = 0.f;
    for (int s2 = 0; s2 < S_; ++s2) c += sl[s2] * enc[((size_t)b * S_ + s2) * H_ + hh];
    sctx[hh] = c;
  }
  __syncthreads();

  float pg = 0.f;
  for (int i = tid; i < H_; i += S_){
    pg += w32[b * H_ + i] * wg_w[i] + h32[b * H_ + i] * wg_w[H_ + i] + sctx[i] * wg_w[2 * H_ + i];
  }
  for (int o = 16; o > 0; o >>= 1) pg += __shfl_xor(pg, o, 32);
  if ((tid & 31) == 0) red[tid >> 5] = pg;
  __syncthreads();
  if (tid == 0){
    float tot = 0.f;
    for (int i = 0; i < 16; ++i) tot += red[i];
    pgen[b] = 1.f / (1.f + expf(-(tot + wg_b[0])));
  }
}

// ---- vocab logits via WMMA, L2-resident f16 embedding; row max via ordered atomicMax ----
__global__ void vocab_mm_kernel(const _Float16* __restrict__ h16, const _Float16* __restrict__ embP,
                                float* __restrict__ logits, unsigned* __restrict__ rowmaxkey){
  int wave = threadIdx.x >> 5;
  int lane = threadIdx.x & 31;
  int tile = blockIdx.x * 8 + wave;
  if (tile >= NT_V) return;                      // wave-uniform: EXEC stays full for WMMA
  v8f c = {};
#pragma unroll 4
  for (int q = 0; q < KSTEPS; ++q){
    v16h a = load_a_frag(h16, q * 32);
    v16h b = load_b_packed(embP, tile, q);
    c = __builtin_amdgcn_wmma_f32_16x16x32_f16(false, a, false, b, (short)0, c, false, false);
  }
  int n = tile * 16 + (lane & 15);
  bool wr = (lane < 16) && (n < V_);             // lanes 0-15: M=r valid batch rows
#pragma unroll
  for (int r = 0; r < 8; ++r){
    if (wr) logits[(size_t)r * VP_ + n] = c[r];
  }
#pragma unroll
  for (int r = 0; r < 8; ++r){
    float v = wr ? c[r] : -3.0e38f;
    for (int o = 16; o > 0; o >>= 1) v = fmaxf(v, __shfl_xor(v, o, 32));
    if (lane == 0) atomicMax(&rowmaxkey[r], fkey(v));
  }
}

// ---- exp(l - max) in place + deterministic per-block partial sums ----
__global__ void exp_sum_kernel(float* __restrict__ logits, const unsigned* __restrict__ rowmaxkey,
                               float* __restrict__ partial){
  int b = blockIdx.y;
  float m = funkey(rowmaxkey[b]);
  int base = blockIdx.x * 2048 + threadIdx.x;
  float s = 0.f;
#pragma unroll
  for (int i = 0; i < 8; ++i){
    int n = base + i * 256;
    if (n < V_){
      float e = expf(logits[(size_t)b * VP_ + n] - m);
      logits[(size_t)b * VP_ + n] = e;
      s += e;
    }
  }
  for (int o = 16; o > 0; o >>= 1) s += __shfl_xor(s, o, 32);
  __shared__ float r2[8];
  if ((threadIdx.x & 31) == 0) r2[threadIdx.x >> 5] = s;
  __syncthreads();
  if (threadIdx.x == 0){
    float t = 0.f;
    for (int i = 0; i < 8; ++i) t += r2[i];
    partial[b * 32 + blockIdx.x] = t;
  }
}

// ---- dense output write: p_gen * softmax(vocab) ----
__global__ void write_out_kernel(const float* __restrict__ expv, const float* __restrict__ partial,
                                 const float* __restrict__ pgen, float* __restrict__ out, int t){
  int b = blockIdx.y;
  float ssum = 0.f;
#pragma unroll
  for (int i = 0; i < 32; ++i) ssum += partial[b * 32 + i];
  float scale = pgen[b] / ssum;
  int j = t / KK_, k = t % KK_;
  size_t obase = (((size_t)b * J_ + j) * KK_ + k) * (size_t)V_;
  int base = blockIdx.x * 2048 + threadIdx.x;
#pragma unroll
  for (int i = 0; i < 8; ++i){
    int n = base + i * 256;
    if (n < V_) out[obase + n] = expv[(size_t)b * VP_ + n] * scale;
  }
}

// ---- pointer-copy scatter: out[b,j,k, x[b,s]] += (1-p_gen)*a_hist[b,s] ----
__global__ void scatter_kernel(const int* __restrict__ x, const float* __restrict__ ahist,
                               const float* __restrict__ pgen, float* __restrict__ out, int t){
  int b = blockIdx.x, s = threadIdx.x;
  int j = t / KK_, k = t % KK_;
  float v = (1.f - pgen[b]) * ahist[b * S_ + s];
  size_t obase = (((size_t)b * J_ + j) * KK_ + k) * (size_t)V_;
  atomicAdd(&out[obase + x[b * S_ + s]], v);
}

extern "C" void kernel_launch(void* const* d_in, const int* in_sizes, int n_in,
                              void* d_out, int out_size, void* d_ws, size_t ws_size,
                              hipStream_t stream){
  (void)in_sizes; (void)n_in; (void)out_size; (void)ws_size;
  const int*   x       = (const int*)  d_in[0];
  const float* dec_in  = (const float*)d_in[1];
  const float* enc     = (const float*)d_in[2];
  const float* hidden  = (const float*)d_in[3];
  const int*   teacher = (const int*)  d_in[4];
  const float* embed   = (const float*)d_in[6];
  const float* w_ih    = (const float*)d_in[7];
  const float* w_hh    = (const float*)d_in[8];
  const float* b_ih    = (const float*)d_in[9];
  const float* b_hh    = (const float*)d_in[10];
  const float* wg_w    = (const float*)d_in[11];
  const float* wg_b    = (const float*)d_in[12];
  float* out = (float*)d_out;

  char* p = (char*)d_ws;
  auto carve = [&](size_t bytes) -> void* {
    void* r = (void*)p;
    p += (bytes + 255) & ~(size_t)255;
    return r;
  };
  _Float16* embP = (_Float16*)carve((size_t)NT_V * KSTEPS * 32 * 16 * 2);  // ~53.8 MB (L2-resident)
  _Float16* wihP = (_Float16*)carve((size_t)NT_G * KSTEPS * 32 * 16 * 2);
  _Float16* whhP = (_Float16*)carve((size_t)NT_G * KSTEPS * 32 * 16 * 2);
  _Float16* h16  = (_Float16*)carve((size_t)16 * H_ * 2);
  _Float16* w16a = (_Float16*)carve((size_t)16 * H_ * 2);
  _Float16* w16b = (_Float16*)carve((size_t)16 * H_ * 2);
  float* h32  = (float*)carve((size_t)B_ * H_ * 4);
  float* w32a = (float*)carve((size_t)B_ * H_ * 4);
  float* w32b = (float*)carve((size_t)B_ * H_ * 4);
  float* gi   = (float*)carve((size_t)B_ * H3_ * 4);
  float* gh   = (float*)carve((size_t)B_ * H3_ * 4);
  float* ahist= (float*)carve((size_t)B_ * S_ * 4);
  float* pgen = (float*)carve((size_t)B_ * 4);
  unsigned* rowmaxkey = (unsigned*)carve((size_t)B_ * 4);
  float* logits  = (float*)carve((size_t)B_ * VP_ * 4);
  float* partial = (float*)carve((size_t)B_ * 32 * 4);

  // one-time: pack weights to f16 B-fragment layout; init h/w state
  pack_b_kernel<<<(NT_V * KSTEPS * 32) / 256, 256, 0, stream>>>(embed, embP, V_, NT_V);
  pack_b_kernel<<<(NT_G * KSTEPS * 32) / 256, 256, 0, stream>>>(w_ih, wihP, H3_, NT_G);
  pack_b_kernel<<<(NT_G * KSTEPS * 32) / 256, 256, 0, stream>>>(w_hh, whhP, H3_, NT_G);
  init_state_kernel<<<(16 * H_) / 256, 256, 0, stream>>>(hidden, dec_in, h32, h16, w32a, w16a, w16b);

  _Float16* w16pp[2] = { w16a, w16b };
  float*    w32pp[2] = { w32a, w32b };

  for (int t = 0; t < J_ * KK_; ++t){
    int cur = t & 1, nxt = cur ^ 1;
    gru_mm_kernel<<<dim3(NT_G / 4, 2), 128, 0, stream>>>(w16pp[cur], h16, wihP, whhP, gi, gh);
    gru_gates_kernel<<<(B_ * H_) / 256, 256, 0, stream>>>(gi, gh, b_ih, b_hh, h32, h16,
                                                          dec_in, embed, teacher,
                                                          w32pp[nxt], w16pp[nxt], t);
    attn_kernel<<<B_, S_, 0, stream>>>(enc, x, h32, w32pp[cur], wg_w, wg_b,
                                       ahist, pgen, rowmaxkey, partial);
    vocab_mm_kernel<<<(NT_V + 7) / 8, 256, 0, stream>>>(h16, embP, logits, rowmaxkey);
    exp_sum_kernel<<<dim3((VP_ + 2047) / 2048, B_), 256, 0, stream>>>(logits, rowmaxkey, partial);
    write_out_kernel<<<dim3((VP_ + 2047) / 2048, B_), 256, 0, stream>>>(logits, partial, pgen, out, t);
    scatter_kernel<<<B_, S_, 0, stream>>>(x, ahist, pgen, out, t);
  }
}
